// FFWD_VQK_79139067396098
// MI455X (gfx1250) — matrix operand order repaired and checked by reading, not compiled, not run on hardware
//
#include <hip/hip_runtime.h>
#include <hip/hip_bf16.h>

#define B_ 4
#define T_ 2048
#define E_ 512
#define H_ 64
#define V_ 32000
#define M_ (B_ * T_)              // 8192 rows total
#define SCALE_ 0.044194173824159216f   // 512^-0.5

typedef __bf16 bf16_t;
typedef bf16_t v8bf  __attribute__((ext_vector_type(8)));
typedef bf16_t v16bf __attribute__((ext_vector_type(16)));
typedef float  v8f   __attribute__((ext_vector_type(8)));

typedef unsigned int u32x4 __attribute__((ext_vector_type(4)));
typedef int          i32x8 __attribute__((ext_vector_type(8)));
typedef int          i32x4 __attribute__((ext_vector_type(4)));

// Tensor Data Mover path (6-arg builtin on this toolchain: g0,g1,g2,g3,g4,cpol)
#if defined(__has_builtin)
#if __has_builtin(__builtin_amdgcn_tensor_load_to_lds) && \
    __has_builtin(__builtin_amdgcn_s_wait_tensorcnt)
#define USE_TDM 1
#endif
#endif
#ifndef USE_TDM
#define USE_TDM 0
#endif

static __device__ __forceinline__ unsigned short f2bf(float f) {
  unsigned int u = __float_as_uint(f);
  u += 0x7fffu + ((u >> 16) & 1u);      // round-to-nearest-even
  return (unsigned short)(u >> 16);
}

static __device__ __forceinline__ v8bf ld8(const unsigned short* p) {
  return *reinterpret_cast<const v8bf*>(p);   // 16B -> global_load_b128 / ds_load_b128
}

static __device__ __forceinline__ v16bf join16(v8bf lo, v8bf hi) {
  v16bf r;
#pragma unroll
  for (int i = 0; i < 8; ++i) { r[i] = lo[i]; r[i + 8] = hi[i]; }
  return r;
}

static __device__ __forceinline__ v8f wmma_bf16(v16bf a, v16bf b, v8f c) {
  // D = A(16x32 bf16) * B(32x16 bf16) + C(16x16 f32)
  return __builtin_amdgcn_wmma_f32_16x16x32_bf16(false, a, false, b, (short)0, c,
                                                 false, false);
}

// A operand 16x32 tile, row-major source, row stride = lda (elements).
// lane<16: row=lane,    K = k0+[0..7]  and k0+[16..23]
// lane>=16: row=lane-16, K = k0+[8..15] and k0+[24..31]
static __device__ __forceinline__ v16bf load_a(const unsigned short* base,
                                               int lr, int lh, int lda, int k0) {
  const unsigned short* p = base + (size_t)lr * lda + k0 + lh * 8;
  return join16(ld8(p), ld8(p + 16));
}

// B operand 32x16 tile: lane holds one column (= one row of the [N,K] weight),
// lane<16: K=k0+[0..7] and k0+[16..23], lane>=16: K=k0+[8..15] and k0+[24..31].
static __device__ __forceinline__ v16bf load_b(const unsigned short* base,
                                               int lr, int lh, int ldb, int k0) {
  const unsigned short* p = base + (size_t)lr * ldb + k0 + lh * 8;
  return join16(ld8(p), ld8(p + 16));
}

// ---------------------------------------------------------------- fp32 -> bf16
__global__ __launch_bounds__(256) void cvt_bf16_kernel(const float* __restrict__ src,
                                                       unsigned short* __restrict__ dst,
                                                       int n4) {
  int i = blockIdx.x * 256 + threadIdx.x;
  if (i >= n4) return;
  float4 f = reinterpret_cast<const float4*>(src)[i];
  ushort4 h; h.x = f2bf(f.x); h.y = f2bf(f.y); h.z = f2bf(f.z); h.w = f2bf(f.w);
  reinterpret_cast<ushort4*>(dst)[i] = h;
}

// ------------------------------------------------------------------ embedding
__global__ __launch_bounds__(256) void embed_kernel(const int* __restrict__ toks,
                                                    const float* __restrict__ emb,
                                                    const float* __restrict__ pos,
                                                    float* __restrict__ xf32,
                                                    unsigned short* __restrict__ xbf) {
  int i = blockIdx.x * 256 + threadIdx.x;          // one float4 per thread
  if (i >= M_ * E_ / 4) return;
  int m  = i >> 7;                                  // row (E_/4 = 128 vec4/row)
  int c4 = (i & 127) * 4;
  int t  = m & (T_ - 1);
  int tok = toks[m];
  float4 e = *reinterpret_cast<const float4*>(emb + (size_t)tok * E_ + c4);
  float4 p = *reinterpret_cast<const float4*>(pos + (size_t)t * E_ + c4);
  float4 x; x.x = e.x + p.x; x.y = e.y + p.y; x.z = e.z + p.z; x.w = e.w + p.w;
  *reinterpret_cast<float4*>(xf32 + (size_t)m * E_ + c4) = x;
  ushort4 h; h.x = f2bf(x.x); h.y = f2bf(x.y); h.z = f2bf(x.z); h.w = f2bf(x.w);
  *reinterpret_cast<ushort4*>(xbf + (size_t)m * E_ + c4) = h;
}

// --------------------------------------------------- q,k projection (N=64 each)
__global__ __launch_bounds__(256) void qk_kernel(const unsigned short* __restrict__ xbf,
                                                 const unsigned short* __restrict__ Wqb,
                                                 const unsigned short* __restrict__ Wkb,
                                                 unsigned short* __restrict__ qbf,
                                                 unsigned short* __restrict__ kbf) {
  int w = threadIdx.x >> 5, lane = threadIdx.x & 31;
  int lr = lane & 15, lh = lane >> 4;
  size_t r0 = (size_t)blockIdx.x * 128 + w * 16;   // 16-row strip per wave
  v8f aq[4], ak[4];
#pragma unroll
  for (int j = 0; j < 4; ++j) { aq[j] = (v8f){}; ak[j] = (v8f){}; }
  const unsigned short* arow = xbf + r0 * E_;
  for (int k0 = 0; k0 < E_; k0 += 32) {
    v16bf a = load_a(arow, lr, lh, E_, k0);
#pragma unroll
    for (int j = 0; j < 4; ++j) {
      aq[j] = wmma_bf16(a, load_b(Wqb + (size_t)(j * 16) * E_, lr, lh, E_, k0), aq[j]);
      ak[j] = wmma_bf16(a, load_b(Wkb + (size_t)(j * 16) * E_, lr, lh, E_, k0), ak[j]);
    }
  }
#pragma unroll
  for (int j = 0; j < 4; ++j)
#pragma unroll
    for (int g = 0; g < 8; ++g) {
      size_t r = r0 + g + lh * 8;
      int n = j * 16 + lr;
      qbf[r * H_ + n] = f2bf(aq[j][g]);
      kbf[r * H_ + n] = f2bf(ak[j][g]);
    }
}

// ---------------------------- v projection, stored transposed: vT[b][c][t] bf16
__global__ __launch_bounds__(256) void vproj_kernel(const unsigned short* __restrict__ xbf,
                                                    const unsigned short* __restrict__ Wvb,
                                                    unsigned short* __restrict__ vT) {
  int w = threadIdx.x >> 5, lane = threadIdx.x & 31;
  int lr = lane & 15, lh = lane >> 4;
  size_t r0 = (size_t)blockIdx.x * 128 + w * 16;
  int n0 = blockIdx.y * 64;
  v8f acc[4];
#pragma unroll
  for (int j = 0; j < 4; ++j) acc[j] = (v8f){};
  const unsigned short* arow = xbf + r0 * E_;
  for (int k0 = 0; k0 < E_; k0 += 32) {
    v16bf a = load_a(arow, lr, lh, E_, k0);
#pragma unroll
    for (int j = 0; j < 4; ++j)
      acc[j] = wmma_bf16(a, load_b(Wvb + (size_t)(n0 + j * 16) * E_, lr, lh, E_, k0),
                         acc[j]);
  }
#pragma unroll
  for (int j = 0; j < 4; ++j)
#pragma unroll
    for (int g = 0; g < 8; ++g) {
      size_t r = r0 + g + lh * 8;
      int b = (int)(r >> 11), t = (int)(r & (T_ - 1));
      int c = n0 + j * 16 + lr;
      vT[((size_t)(b * E_) + c) * T_ + t] = f2bf(acc[j][g]);
    }
}

// -------------------- attention: one (batch, 16-row tile) per workgroup, 8 waves
// LDS: S = f32 scores [16][2048] (128KB), P = bf16 probs [16][2048] (64KB)
__global__ __launch_bounds__(256) void attn_kernel(const unsigned short* __restrict__ qbf,
                                                   const unsigned short* __restrict__ kbf,
                                                   const unsigned short* __restrict__ vT,
                                                   const float* __restrict__ xf32,
                                                   unsigned short* __restrict__ xobf) {
  extern __shared__ char smem[];
  float*          S = reinterpret_cast<float*>(smem);                  // 16x2048 f32
  unsigned short* P = reinterpret_cast<unsigned short*>(smem + 16 * T_ * 4);

  int tid = threadIdx.x;
  int w = tid >> 5, lane = tid & 31;
  int lr = lane & 15, lh = lane >> 4;
  int b = blockIdx.y;
  int m0 = blockIdx.x * 16;
  size_t r0 = (size_t)b * T_ + m0;
  int lim = (m0 + 16 + 31) & ~31;                 // K extent for the PV loop

  // ---- phase A: S = (Q Kt) * scale for this wave's 256-column span
  const unsigned short* qrow = qbf + r0 * H_;
  v16bf aq0 = load_a(qrow, lr, lh, H_, 0);
  v16bf aq1 = load_a(qrow, lr, lh, H_, 32);
  for (int j = 0; j < 16; ++j) {
    int s0 = w * 256 + j * 16;
    if (s0 > m0 + 15) break;                      // fully masked tile
    const unsigned short* krow = kbf + ((size_t)b * T_ + s0) * H_;
    v8f acc = (v8f){};
    acc = wmma_bf16(aq0, load_b(krow, lr, lh, H_, 0),  acc);
    acc = wmma_bf16(aq1, load_b(krow, lr, lh, H_, 32), acc);
#pragma unroll
    for (int g = 0; g < 8; ++g)
      S[(g + lh * 8) * T_ + s0 + lr] = acc[g] * SCALE_;
  }
  __syncthreads();

  // ---- softmax: 16 threads per row (half-wave), causal mask via scan limit
  int row = tid >> 4;                             // 0..15
  int j16 = tid & 15;
  int len = m0 + row + 1;                         // valid columns for this row
  float* Sr = S + row * T_;
  float mx = -3.0e38f;
  for (int i = j16; i < len; i += 16) mx = fmaxf(mx, Sr[i]);
#pragma unroll
  for (int d = 8; d >= 1; d >>= 1) mx = fmaxf(mx, __shfl_xor(mx, d, 32));
  float sum = 0.f;
  for (int i = j16; i < len; i += 16) { float e = __expf(Sr[i] - mx); Sr[i] = e; sum += e; }
#pragma unroll
  for (int d = 8; d >= 1; d >>= 1) sum += __shfl_xor(sum, d, 32);
  float rinv = 1.f / sum;
  unsigned short* Pr = P + row * T_;
  for (int i = j16; i < lim; i += 16)
    Pr[i] = (i < len) ? f2bf(Sr[i] * rinv) : (unsigned short)0;
  __syncthreads();

  // ---- phase B: out = P @ V (A from LDS, B from transposed vT), + residual
  int c0 = w * 64;                                // this wave's 64 output cols
  v8f acc[4];
#pragma unroll
  for (int j = 0; j < 4; ++j) acc[j] = (v8f){};
  for (int s0 = 0; s0 < lim; s0 += 32) {
    const unsigned short* pp = P + (size_t)lr * T_ + s0 + lh * 8;
    v16bf ap = join16(ld8(pp), ld8(pp + 16));     // ds_load_b128 x2
#pragma unroll
    for (int j = 0; j < 4; ++j) {
      const unsigned short* vp =
          vT + ((size_t)(b * E_) + c0 + j * 16 + lr) * T_ + s0 + lh * 8;
      acc[j] = wmma_bf16(ap, join16(ld8(vp), ld8(vp + 16)), acc[j]);
    }
  }
#pragma unroll
  for (int j = 0; j < 4; ++j)
#pragma unroll
    for (int g = 0; g < 8; ++g) {
      size_t idx = (r0 + g + lh * 8) * E_ + c0 + j * 16 + lr;
      xobf[idx] = f2bf(xf32[idx] + acc[j][g]);
    }
}

// -------------------------------- logits: 128x128 tile per WG, 8 waves, K=512
// B operand (Wout tile) staged into LDS by the Tensor Data Mover, double-buffered
// with K=64 per descriptor (16 WMMAs between sync points): wave 0 issues
// tensor_load_to_lds + s_wait_tensorcnt, everyone else just barriers.
__global__ __launch_bounds__(256) void logits_kernel(const unsigned short* __restrict__ xobf,
                                                     const unsigned short* __restrict__ Woutb,
                                                     const float* __restrict__ bout,
                                                     float* __restrict__ out) {
  int w = threadIdx.x >> 5, lane = threadIdx.x & 31;
  int lr = lane & 15, lh = lane >> 4;
  size_t r0 = (size_t)blockIdx.x * 128 + w * 16;
  int n0 = blockIdx.y * 128;
  v8f acc[8];
#pragma unroll
  for (int j = 0; j < 8; ++j) acc[j] = (v8f){};
  const unsigned short* arow = xobf + r0 * E_;

#if USE_TDM
  __shared__ unsigned short Bb[2][128 * 64];      // double-buffered 128x64 bf16 tile
  const unsigned lds0 = (unsigned)(size_t)&Bb[0][0];  // LDS byte address (low 32 bits)
  const unsigned lds1 = (unsigned)(size_t)&Bb[1][0];

  // D# builder: 2-D tile (tile_dim0=64 along K, tile_dim1=128 rows), bf16 elements.
  auto tdm_issue = [&](unsigned lds_addr, int k0) {
    unsigned long long ga =
        (unsigned long long)(size_t)(Woutb + (size_t)n0 * E_ + k0);
    u32x4 g0;
    g0.x = 0x1u;                                   // count=1 valid descriptor
    g0.y = lds_addr;                               // D#.lds_addr
    g0.z = (unsigned)(ga & 0xffffffffu);           // global_addr[31:0]
    g0.w = (unsigned)((ga >> 32) & 0x01ffffffu)    // global_addr[56:32]
           | 0x80000000u;                          // type=2 ("image")
    i32x8 g1;
    g1[0] = 0x10000;                               // data_size=1 -> 2 bytes/elem
    g1[1] = (int)(((unsigned)E_ & 0xffffu) << 16); // tensor_dim0[15:0] = 512
    g1[2] = (int)((((unsigned)E_ >> 16) & 0xffffu) |
                  (((unsigned)V_ & 0xffffu) << 16));   // dim0 hi | tensor_dim1 lo
    g1[3] = (int)(((((unsigned)V_ >> 16) & 0xffffu)) |
                  (64u << 16));                    // dim1 hi | tile_dim0 = 64
    g1[4] = 128;                                   // tile_dim1 = 128, tile_dim2 = 0
    g1[5] = E_;                                    // tensor_dim0_stride = 512
    g1[6] = 0;
    g1[7] = 0;
    i32x4 z4 = {0, 0, 0, 0};                       // 2-D: groups 2/3 unused
    i32x8 z8 = {0, 0, 0, 0, 0, 0, 0, 0};           // extra group (6-arg toolchain)
    __builtin_amdgcn_tensor_load_to_lds(g0, g1, z4, z4, z8, 0);
  };

  const int nch = E_ / 64;                         // 8 K-chunks of 64
  if (w == 0) tdm_issue(lds0, 0);
  for (int i = 0; i < nch; ++i) {
    int k0 = i * 64;
    if (w == 0) {
      if (i + 1 < nch) {
        tdm_issue(((i + 1) & 1) ? lds1 : lds0, k0 + 64);
        __builtin_amdgcn_s_wait_tensorcnt(1);      // chunk i landed, i+1 in flight
      } else {
        __builtin_amdgcn_s_wait_tensorcnt(0);
      }
    }
    __syncthreads();                               // buf[i&1] visible to all waves
    const unsigned short* Bcur = (i & 1) ? &Bb[1][0] : &Bb[0][0];
    if (k0 + 64 < E_)
      __builtin_prefetch(arow + (size_t)lr * E_ + k0 + 64, 0, 0);  // global_prefetch
    v16bf a0 = load_a(arow, lr, lh, E_, k0);
    v16bf a1 = load_a(arow, lr, lh, E_, k0 + 32);
#pragma unroll
    for (int j = 0; j < 8; ++j) {                  // B frags via ds_load_b128
      const unsigned short* brow = Bcur + (size_t)(j * 16) * 64;
      acc[j] = wmma_bf16(a0, load_b(brow, lr, lh, 64, 0),  acc[j]);
      acc[j] = wmma_bf16(a1, load_b(brow, lr, lh, 64, 32), acc[j]);
    }
    __syncthreads();                               // done reading before overwrite
  }
#else
  for (int k0 = 0; k0 < E_; k0 += 32) {
    if (k0 + 32 < E_)
      __builtin_prefetch(arow + (size_t)lr * E_ + k0 + 32, 0, 0);
    v16bf a = load_a(arow, lr, lh, E_, k0);
#pragma unroll
    for (int j = 0; j < 8; ++j)
      acc[j] = wmma_bf16(a, load_b(Woutb + (size_t)(n0 + j * 16) * E_, lr, lh, E_, k0),
                         acc[j]);
  }
#endif

#pragma unroll
  for (int j = 0; j < 8; ++j) {
    int n = n0 + j * 16 + lr;
    float bias = bout[n];
#pragma unroll
    for (int g = 0; g < 8; ++g)
      out[(r0 + g + lh * 8) * V_ + n] = acc[j][g] + bias;
  }
}

extern "C" void kernel_launch(void* const* d_in, const int* in_sizes, int n_in,
                              void* d_out, int out_size, void* d_ws, size_t ws_size,
                              hipStream_t stream) {
  (void)in_sizes; (void)n_in; (void)out_size; (void)ws_size;
  const int*   toks = (const int*)  d_in[0];
  const float* emb  = (const float*)d_in[1];
  const float* pos  = (const float*)d_in[2];
  const float* Wq   = (const float*)d_in[3];
  const float* Wk   = (const float*)d_in[4];
  const float* Wv   = (const float*)d_in[5];
  const float* Wout = (const float*)d_in[6];
  const float* bout = (const float*)d_in[7];
  float* out = (float*)d_out;

  char* ws = (char*)d_ws;
  size_t off = 0;
  auto take = [&](size_t bytes) -> char* {
    char* p = ws + off;
    off = (off + bytes + 255) & ~(size_t)255;
    return p;
  };
  float*          xf32  = (float*)         take((size_t)M_ * E_ * 4);
  unsigned short* xbf   = (unsigned short*)take((size_t)M_ * E_ * 2);
  unsigned short* qbf   = (unsigned short*)take((size_t)M_ * H_ * 2);
  unsigned short* kbf   = (unsigned short*)take((size_t)M_ * H_ * 2);
  unsigned short* vT    = (unsigned short*)take((size_t)M_ * E_ * 2);
  unsigned short* xobf  = (unsigned short*)take((size_t)M_ * E_ * 2);
  unsigned short* Wqb   = (unsigned short*)take((size_t)H_ * E_ * 2);
  unsigned short* Wkb   = (unsigned short*)take((size_t)H_ * E_ * 2);
  unsigned short* Wvb   = (unsigned short*)take((size_t)E_ * E_ * 2);
  unsigned short* Woutb = (unsigned short*)take((size_t)V_ * E_ * 2);

  // one-time (per launch) weight conversion to bf16
  auto cvt = [&](const float* s, unsigned short* d, int n) {
    int n4 = n / 4;
    cvt_bf16_kernel<<<(n4 + 255) / 256, 256, 0, stream>>>(s, d, n4);
  };
  cvt(Wq,   Wqb,   H_ * E_);
  cvt(Wk,   Wkb,   H_ * E_);
  cvt(Wv,   Wvb,   E_ * E_);
  cvt(Wout, Woutb, V_ * E_);

  embed_kernel<<<(M_ * E_ / 4 + 255) / 256, 256, 0, stream>>>(toks, emb, pos, xf32, xbf);
  qk_kernel<<<M_ / 128, 256, 0, stream>>>(xbf, Wqb, Wkb, qbf, kbf);
  vproj_kernel<<<dim3(M_ / 128, E_ / 64), 256, 0, stream>>>(xbf, Wvb, vT);

  const int attn_smem = 16 * T_ * 4 + 16 * T_ * 2;   // 192 KB (<= 320 KB/WGP)
  (void)hipFuncSetAttribute((const void*)attn_kernel,
                            hipFuncAttributeMaxDynamicSharedMemorySize, attn_smem);
  attn_kernel<<<dim3(T_ / 16, B_), 256, attn_smem, stream>>>(qbf, kbf, vT, xf32, xobf);

  logits_kernel<<<dim3(M_ / 128, V_ / 128), 256, 0, stream>>>(xobf, Woutb, bout, out);
}
